// ConditionedRecurrentEdgeModel_37177236914588
// MI455X (gfx1250) — matrix operand order repaired and checked by reading, not compiled, not run on hardware
//
#include <hip/hip_runtime.h>

typedef __attribute__((ext_vector_type(16))) __bf16 v16bf;
typedef __attribute__((ext_vector_type(8)))  __bf16 v8bf;
typedef __attribute__((ext_vector_type(8)))  float  v8f;

#define LN_EPS 1e-5f

// ---------------- WMMA helpers ----------------
static __device__ __forceinline__ v8f wmma_bf(v16bf a, v16bf b, v8f c) {
    return __builtin_amdgcn_wmma_f32_16x16x32_bf16(false, a, false, b, (short)0, c, false, false);
}

// A-operand pack (16-bit A 16x32 layout): lane holds 8 contiguous K at p[0..7]
// and 8 contiguous K at p[16..23] (caller applies the +8 offset for lanes>=16).
static __device__ __forceinline__ v16bf packA_f32(const float* p) {
    v8f f0 = *(const v8f*)(p);
    v8f f1 = *(const v8f*)(p + 16);
    v16bf a;
#pragma unroll
    for (int j = 0; j < 8; ++j) { a[j] = (__bf16)f0[j]; a[j + 8] = (__bf16)f1[j]; }
    return a;
}
static __device__ __forceinline__ v16bf packA_bf16(const __bf16* p) {
    v8bf lo = *(const v8bf*)(p);
    v8bf hi = *(const v8bf*)(p + 16);
    return __builtin_shufflevector(lo, hi, 0,1,2,3,4,5,6,7,8,9,10,11,12,13,14,15);
}

static __device__ __forceinline__ float red16(float x) {
    x += __shfl_xor(x, 1, 32);
    x += __shfl_xor(x, 2, 32);
    x += __shfl_xor(x, 4, 32);
    x += __shfl_xor(x, 8, 32);
    return x;
}
static __device__ __forceinline__ float sigm(float x) { return 1.f / (1.f + __expf(-x)); }
static __device__ __forceinline__ float tanh_fast(float x) { return 1.f - 2.f / (__expf(2.f * x) + 1.f); }

// ---------------- weight prep: f32 [N,K] row-major -> bf16 WMMA B layout ----
// B entry layout (v16bf units): [(kchunk*NT + ntile)*32 + lane], lane = khalf*16 + n,
// element j -> K = kchunk*32 + khalf*8 + (j&7) + (j>>3)*16, column N = ntile*16 + n.
static __device__ __forceinline__ void conv_b(const float* W, int K, int NT, __bf16* dst, int idx) {
    int j    = idx & 15;
    int lane = (idx >> 4) & 31;
    int t    = idx >> 9;
    int nt   = t % NT;
    int kc   = t / NT;
    int n = nt * 16 + (lane & 15);
    int k = kc * 32 + ((lane >> 4) << 3) + (j & 7) + ((j >> 3) << 4);
    dst[idx] = (__bf16)W[n * K + k];
}

#define NB_W1  (12 * 8  * 512)   // K=384 -> 12 chunks, N=128 -> 8 tiles
#define NB_WIH (4  * 24 * 512)   // K=128 -> 4 chunks,  N=384 -> 24 tiles
#define NB_WHH (4  * 24 * 512)
#define NB_WX  (4  * 4  * 512)   // K=128 -> 4 chunks,  N=64  -> 4 tiles
#define OFF_WIH (NB_W1)
#define OFF_WHH (NB_W1 + NB_WIH)
#define OFF_WX  (NB_W1 + NB_WIH + NB_WHH)
#define NB_TOTAL (NB_W1 + NB_WIH + NB_WHH + NB_WX)

__global__ void prep_weights(const float* W1, const float* Wih, const float* Whh,
                             const float* Wx, __bf16* ws) {
    int i = blockIdx.x * blockDim.x + threadIdx.x;
    int stride = gridDim.x * blockDim.x;
    for (; i < NB_TOTAL; i += stride) {
        if (i < OFF_WIH)      conv_b(W1,  384, 8,  ws,            i);
        else if (i < OFF_WHH) conv_b(Wih, 128, 24, ws + OFF_WIH,  i - OFF_WIH);
        else if (i < OFF_WX)  conv_b(Whh, 128, 24, ws + OFF_WHH,  i - OFF_WHH);
        else                  conv_b(Wx,  128, 4,  ws + OFF_WX,   i - OFF_WX);
    }
}

// ---------------- main fused kernel ----------------
// Block = 128 threads = 4 waves; each wave owns 16 edges.
// No min-waves bound: letting RA use >256 VGPRs (3 waves/SIMD) is spill-free,
// which beats the spilling 256-VGPR schedules (rounds 2-4 all spilled in the
// GEMM1 inner stream behind full loadcnt drains).
__global__ __launch_bounds__(128)
void edge_model_kernel(const float* __restrict__ src, const float* __restrict__ dst_n,
                       const float* __restrict__ ea,  const float* __restrict__ h,
                       const float* __restrict__ u,   const int*   __restrict__ batch,
                       const float* __restrict__ winding, const float* __restrict__ Ww,
                       const float* __restrict__ bw,  const float* __restrict__ gw,
                       const float* __restrict__ betaw,
                       const float* __restrict__ b1,  const float* __restrict__ g1,
                       const float* __restrict__ beta1,
                       const float* __restrict__ bih, const float* __restrict__ bhh,
                       const float* __restrict__ bx,
                       const __bf16* __restrict__ ws, float* __restrict__ out, int E) {
    __shared__ float s_par[1600];                                   // 6.25 KB
    __shared__ __attribute__((aligned(16))) __bf16 s_bufA[4][2048]; // 16 KB  (w / x1 bf16, x_out f32 bounce)
    __shared__ __attribute__((aligned(16))) float  s_bufH[4][2048]; // 32 KB  (h -> h_new, in place)

    const int tid = threadIdx.x;
    // cooperative param load
    for (int i = tid; i < 1600; i += 128) {
        float v;
        if      (i < 128)  v = bw[i];
        else if (i < 256)  v = gw[i - 128];
        else if (i < 384)  v = betaw[i - 256];
        else if (i < 512)  v = b1[i - 384];
        else if (i < 640)  v = g1[i - 512];
        else if (i < 768)  v = beta1[i - 640];
        else if (i < 1152) v = bih[i - 768];
        else if (i < 1536) v = bhh[i - 1152];
        else               v = bx[i - 1536];
        s_par[i] = v;
    }
    __syncthreads();

    const float* pbw    = s_par;
    const float* pgw    = s_par + 128;
    const float* pbetaw = s_par + 256;
    const float* pb1    = s_par + 384;
    const float* pg1    = s_par + 512;
    const float* pbeta1 = s_par + 640;
    const float* pbih   = s_par + 768;
    const float* pbhh   = s_par + 1152;
    const float* pbx    = s_par + 1536;

    const int wave = tid >> 5;
    const int lane = tid & 31;
    const int ln   = lane & 15;
    const int hi   = lane >> 4;      // K-half select for A operands
    const int off  = hi * 8;

    const int base = (blockIdx.x * 4 + wave) * 16;
    if (base >= E) return;

    __bf16* bufA = s_bufA[wave];
    float*  bufH = s_bufH[wave];

    // ---- stage h tile (coalesced) ----
    for (int i = lane; i < 2048; i += 32) {
        int m = i >> 7, e = base + m;
        bufH[i] = (e < E) ? h[(long)e * 128 + (i & 127)] : 0.f;
    }

    // ---- winding MLP + LN + ReLU -> bufA (bf16, row-major [16][128]) ----
    float wwa[8], wwb[8];
#pragma unroll
    for (int nt = 0; nt < 8; ++nt) {
        int c = nt * 16 + ln;
        wwa[nt] = Ww[2 * c];
        wwb[nt] = Ww[2 * c + 1];
    }
#pragma unroll
    for (int r = 0; r < 8; ++r) {
        int m = r + hi * 8;
        int e = base + m; if (e >= E) e = E - 1;
        float w0 = winding[2 * e], w1 = winding[2 * e + 1];
        float v[8], s = 0.f, q = 0.f;
#pragma unroll
        for (int nt = 0; nt < 8; ++nt) {
            int c = nt * 16 + ln;
            float x = w0 * wwa[nt] + w1 * wwb[nt] + pbw[c];
            v[nt] = x; s += x; q += x * x;
        }
        s = red16(s); q = red16(q);
        float mu = s * (1.f / 128.f);
        float rs = rsqrtf(q * (1.f / 128.f) - mu * mu + LN_EPS);
#pragma unroll
        for (int nt = 0; nt < 8; ++nt) {
            int c = nt * 16 + ln;
            float x = fmaxf((v[nt] - mu) * rs * pgw[c] + pbetaw[c], 0.f);
            bufA[m * 128 + c] = (__bf16)x;
        }
    }

    // ---- GEMM1: x[16,384] @ W1^T -> acc[8] tiles of [16,16] ----
    const v16bf* B1 = (const v16bf*)ws;
    int eA = base + ln; if (eA >= E) eA = E - 1;
    int bi = batch[eA];
    v8f acc[8] = {};
#pragma unroll
    for (int k = 0; k < 12; ++k) {
        v16bf a;
        if      (k < 2)  a = packA_f32(src   + (long)eA * 64 + k * 32 + off);
        else if (k < 4)  a = packA_f32(dst_n + (long)eA * 64 + (k - 2) * 32 + off);
        else if (k < 6)  a = packA_f32(ea    + (long)eA * 64 + (k - 4) * 32 + off);
        else if (k < 8)  a = packA_f32(u     + (long)bi * 64 + (k - 6) * 32 + off);
        else             a = packA_bf16(bufA + ln * 128 + (k - 8) * 32 + off);
#pragma unroll
        for (int nt = 0; nt < 8; ++nt)
            acc[nt] = wmma_bf(a, B1[(k * 8 + nt) * 32 + lane], acc[nt]);
    }

    // ---- LN + ReLU on mlp1 output (D layout), restage as bf16 A source ----
#pragma unroll
    for (int r = 0; r < 8; ++r) {
        int m = r + hi * 8;
        float v[8], s = 0.f, q = 0.f;
#pragma unroll
        for (int nt = 0; nt < 8; ++nt) {
            int c = nt * 16 + ln;
            float x = acc[nt][r] + pb1[c];
            v[nt] = x; s += x; q += x * x;
        }
        s = red16(s); q = red16(q);
        float mu = s * (1.f / 128.f);
        float rs = rsqrtf(q * (1.f / 128.f) - mu * mu + LN_EPS);
#pragma unroll
        for (int nt = 0; nt < 8; ++nt) {
            int c = nt * 16 + ln;
            float x = fmaxf((v[nt] - mu) * rs * pg1[c] + pbeta1[c], 0.f);
            bufA[m * 128 + c] = (__bf16)x;
        }
    }

    // ---- GRU: preload A operands, then per-channel-tile gate GEMMs ----
    v16bf Ax[4], Ah[4];
#pragma unroll
    for (int k = 0; k < 4; ++k) {
        Ax[k] = packA_bf16(bufA + ln * 128 + k * 32 + off);
        Ah[k] = packA_f32 (bufH + ln * 128 + k * 32 + off);
    }
    const v16bf* Bih = (const v16bf*)(ws + OFF_WIH);
    const v16bf* Bhh = (const v16bf*)(ws + OFF_WHH);

    for (int ct = 0; ct < 8; ++ct) {
        v8f ar = {}, az = {}, ain = {}, ahn = {};
#pragma unroll
        for (int k = 0; k < 4; ++k) {
            ar  = wmma_bf(Ax[k], Bih[(k * 24 + ct)      * 32 + lane], ar);
            ar  = wmma_bf(Ah[k], Bhh[(k * 24 + ct)      * 32 + lane], ar);
            az  = wmma_bf(Ax[k], Bih[(k * 24 + 8 + ct)  * 32 + lane], az);
            az  = wmma_bf(Ah[k], Bhh[(k * 24 + 8 + ct)  * 32 + lane], az);
            ain = wmma_bf(Ax[k], Bih[(k * 24 + 16 + ct) * 32 + lane], ain);
            ahn = wmma_bf(Ah[k], Bhh[(k * 24 + 16 + ct) * 32 + lane], ahn);
        }
        int c = ct * 16 + ln;
        float br = pbih[c] + pbhh[c];
        float bz = pbih[128 + c] + pbhh[128 + c];
        float bin = pbih[256 + c], bhn = pbhh[256 + c];
#pragma unroll
        for (int r = 0; r < 8; ++r) {
            int m = r + hi * 8;
            float rr = sigm(ar[r] + br);
            float zz = sigm(az[r] + bz);
            float nn = tanh_fast(ain[r] + bin + rr * (ahn[r] + bhn));
            float ho = bufH[m * 128 + c];
            bufH[m * 128 + c] = (1.f - zz) * nn + zz * ho;   // h_new, in place
        }
    }

    // ---- write h_new (coalesced) ----
    float* hout = out + (long)E * 64;
    for (int i = lane; i < 2048; i += 32) {
        int m = i >> 7, e = base + m;
        if (e < E) hout[(long)e * 128 + (i & 127)] = bufH[i];
    }

    // ---- mlp_x: h_new[16,128] @ Wx^T -> x_out[16,64] ----
    v16bf An[4];
#pragma unroll
    for (int k = 0; k < 4; ++k) An[k] = packA_f32(bufH + ln * 128 + k * 32 + off);
    const v16bf* Bx = (const v16bf*)(ws + OFF_WX);
    v8f ao[4] = {};
#pragma unroll
    for (int nt = 0; nt < 4; ++nt)
#pragma unroll
        for (int k = 0; k < 4; ++k)
            ao[nt] = wmma_bf(An[k], Bx[(k * 4 + nt) * 32 + lane], ao[nt]);

    float* bufO = (float*)bufA;  // reuse 4 KB as [16][64] f32 bounce
#pragma unroll
    for (int nt = 0; nt < 4; ++nt) {
        int c = nt * 16 + ln;
        float bb = pbx[c];
#pragma unroll
        for (int r = 0; r < 8; ++r) {
            int m = r + hi * 8;
            bufO[m * 64 + c] = ao[nt][r] + bb;
        }
    }
    for (int i = lane; i < 1024; i += 32) {
        int m = i >> 6, e = base + m;
        if (e < E) out[(long)e * 64 + (i & 63)] = bufO[i];
    }
}

extern "C" void kernel_launch(void* const* d_in, const int* in_sizes, int n_in,
                              void* d_out, int out_size, void* d_ws, size_t ws_size,
                              hipStream_t stream) {
    const float* src     = (const float*)d_in[0];
    const float* dst_n   = (const float*)d_in[1];
    const float* ea      = (const float*)d_in[2];
    const float* h       = (const float*)d_in[3];
    const float* u       = (const float*)d_in[4];
    const int*   batch   = (const int*)  d_in[5];
    const float* winding = (const float*)d_in[6];
    const float* Ww      = (const float*)d_in[7];
    const float* bw      = (const float*)d_in[8];
    const float* gw      = (const float*)d_in[9];
    const float* betaw   = (const float*)d_in[10];
    const float* W1      = (const float*)d_in[11];
    const float* b1      = (const float*)d_in[12];
    const float* g1      = (const float*)d_in[13];
    const float* beta1   = (const float*)d_in[14];
    const float* Wih     = (const float*)d_in[15];
    const float* Whh     = (const float*)d_in[16];
    const float* bih     = (const float*)d_in[17];
    const float* bhh     = (const float*)d_in[18];
    const float* Wx      = (const float*)d_in[19];
    const float* bx      = (const float*)d_in[20];

    int E = in_sizes[0] / 64;
    __bf16* ws = (__bf16*)d_ws;

    prep_weights<<<(NB_TOTAL + 255) / 256, 256, 0, stream>>>(W1, Wih, Whh, Wx, ws);

    int blocks = (E + 63) / 64;
    edge_model_kernel<<<blocks, 128, 0, stream>>>(
        src, dst_n, ea, h, u, batch, winding, Ww, bw, gw, betaw,
        b1, g1, beta1, bih, bhh, bx, ws, (float*)d_out, E);
}
